// PerlinAttention_45440753991786
// MI455X (gfx1250) — matrix operand order repaired and checked by reading, not compiled, not run on hardware
//
#include <hip/hip_runtime.h>
#include <hip/hip_bf16.h>
#include <math.h>

// Problem constants (fixed by reference)
#define NB   4
#define HH   12
#define TT   1024
#define HIDD 64
#define FF   266      // performer features
#define FP   288      // padded feature dim (multiple of 32 for WMMA K)
#define TMD  128
#define NH   48       // NB*HH
#define NHT  49152    // NB*HH*TT
#define TGT  98304    // top-k target per batch: kk(=8)*T*H

typedef __bf16 bf16_t;
typedef bf16_t v16bf __attribute__((ext_vector_type(16)));
typedef float  v8f   __attribute__((ext_vector_type(8)));

__device__ inline unsigned short f2bf(float f) {
  unsigned int x = __builtin_bit_cast(unsigned int, f);
  x += 0x7fffu + ((x >> 16) & 1u);          // round-to-nearest-even
  return (unsigned short)(x >> 16);
}
__device__ inline float bf2f(unsigned short u) {
  unsigned int x = ((unsigned int)u) << 16;
  return __builtin_bit_cast(float, x);
}

union U4   { uint4 u; unsigned short s[8]; };
union Frag { uint4 u[2]; v16bf v; };

// ---------------------------------------------------------------------------
// Generic bf16 WMMA GEMM: C[M,N] = A[M,K] * B[K,N] (+bias), row-major, batched.
// Block tile 64x128, 256 threads = 8 waves, each wave 16(M)x64(N) via 4 accums.
// LDS: A row-major [64][40-pad], B TRANSPOSED [128][40-pad] so every WMMA
// fragment is two contiguous 16B runs -> ds_load_b128 (no scalar gathers).
// Requires K % 32 == 0 and N % 8 == 0 (true for all uses here).
// ---------------------------------------------------------------------------
__global__ __launch_bounds__(256)
void k_gemm_bf16(const unsigned short* __restrict__ A,
                 const unsigned short* __restrict__ B,
                 float* __restrict__ C,
                 int M, int Nn, int K,
                 long long sA, long long sB, long long sC,
                 const float* __restrict__ bias) {
  __shared__ unsigned short As[64 * 40];    // [m][k], row stride 40
  __shared__ unsigned short Bs[128 * 40];   // [n][k] (transposed), row stride 40
  const int tid  = threadIdx.x;
  const int lane = tid & 31;
  const int w    = tid >> 5;
  const int wm   = w >> 1;        // 0..3
  const int wn   = w & 1;         // 0..1
  const int m0   = blockIdx.y * 64;
  const int n0   = blockIdx.x * 128;
  const long long bA = (long long)blockIdx.z * sA;
  const long long bB = (long long)blockIdx.z * sB;
  const long long bC = (long long)blockIdx.z * sC;

  v8f acc[4];
  #pragma unroll
  for (int i = 0; i < 4; ++i) acc[i] = 0.0f;

  const int mloc  = wm * 16 + (lane & 15);
  const int khalf = lane >> 4;

  // per-thread staging coordinates
  const int amm = tid >> 2;             // A: row 0..63
  const int ack = (tid & 3) << 3;       // A: k-chunk 0/8/16/24
  const int agm = m0 + amm;

  for (int k0 = 0; k0 < K; k0 += 32) {
    // ---- stage A tile: one 16B chunk per thread ----
    {
      U4 val; val.u = make_uint4(0u, 0u, 0u, 0u);
      if (agm < M)
        val.u = *(const uint4*)(A + bA + (long long)agm * K + (k0 + ack));
      *(uint4*)(&As[amm * 40 + ack]) = val.u;
      if (k0 + 32 < K && agm < M)
        __builtin_prefetch(A + bA + (long long)agm * K + (k0 + 32 + ack), 0, 1);
    }
    // ---- stage B tile: two 16B chunks per thread, transposed into LDS ----
    #pragma unroll
    for (int i = 0; i < 2; ++i) {
      int chunk = tid + (i << 8);       // 0..511
      int kk = chunk >> 4;              // 0..31
      int cn = (chunk & 15) << 3;       // 0..120
      int gn = n0 + cn;
      U4 val; val.u = make_uint4(0u, 0u, 0u, 0u);
      if (gn < Nn)
        val.u = *(const uint4*)(B + bB + (long long)(k0 + kk) * Nn + gn);
      #pragma unroll
      for (int c = 0; c < 8; ++c) Bs[(cn + c) * 40 + kk] = val.s[c];
      if (k0 + 32 < K && gn < Nn)
        __builtin_prefetch(B + bB + (long long)(k0 + 32 + kk) * Nn + gn, 0, 1);
    }
    __syncthreads();

    // A fragment (16x32): lanes 0-15 M=0..15 K0-7/16-23; lanes 16-31 K8-15/24-31
    Frag fa;
    fa.u[0] = *(const uint4*)(&As[mloc * 40 + (khalf << 3)]);
    fa.u[1] = *(const uint4*)(&As[mloc * 40 + 16 + (khalf << 3)]);
    #pragma unroll
    for (int nt = 0; nt < 4; ++nt) {
      // B fragment (32x16): lanes 0-15 K0-7/16-23; lanes 16-31 K8-15/24-31, from Bs[n][k]
      int nloc = wn * 64 + nt * 16 + (lane & 15);
      Frag fb;
      fb.u[0] = *(const uint4*)(&Bs[nloc * 40 + (khalf << 3)]);
      fb.u[1] = *(const uint4*)(&Bs[nloc * 40 + 16 + (khalf << 3)]);
      acc[nt] = __builtin_amdgcn_wmma_f32_16x16x32_bf16(
          false, fa.v, false, fb.v, (short)0, acc[nt], false, false);
    }
    __syncthreads();
  }

  // C layout: lane = col + 16*(m>=8); VGPR r -> row m = r (+8)
  const int ccol = lane & 15;
  const int mhi  = (lane >> 4) & 1;
  for (int nt = 0; nt < 4; ++nt) {
    int gn = n0 + wn * 64 + nt * 16 + ccol;
    float bv = (bias != nullptr && gn < Nn) ? bias[gn] : 0.0f;
    #pragma unroll
    for (int r = 0; r < 8; ++r) {
      int gm = m0 + wm * 16 + mhi * 8 + r;
      if (gm < M && gn < Nn)
        C[bC + (long long)gm * Nn + gn] = acc[nt][r] + bv;
    }
  }
}

// ---------------------------------------------------------------------------
// Weight conversions: pm^T (dn folded) padded to [64,FP]; enc_w/dec_w -> bf16
// ---------------------------------------------------------------------------
__global__ void k_convert_weights(const float* __restrict__ pm,
                                  const float* __restrict__ encw,
                                  const float* __restrict__ decw,
                                  unsigned short* __restrict__ pmT,
                                  unsigned short* __restrict__ encw_bf,
                                  unsigned short* __restrict__ decw_bf) {
  const float dn = 0.35355339059327373f;  // 64^-0.25
  int idx = blockIdx.x * blockDim.x + threadIdx.x;
  const int npm = 64 * FP;
  if (idx < npm) {
    int k = idx / FP, f = idx % FP;
    float v = (f < FF) ? dn * pm[f * 64 + k] : 0.0f;
    pmT[idx] = f2bf(v);
  } else if (idx < npm + 16384) {
    int j = idx - npm;
    encw_bf[j] = f2bf(encw[j]);
  } else if (idx < npm + 32768) {
    int j = idx - npm - 16384;
    decw_bf[j] = f2bf(decw[j]);
  }
}

__global__ void k_convert_x(const float* __restrict__ q,
                            const float* __restrict__ k,
                            unsigned short* __restrict__ qb,
                            unsigned short* __restrict__ kb) {
  long long idx = (long long)blockIdx.x * blockDim.x + threadIdx.x;
  const long long n = (long long)NHT * HIDD;
  if (idx < n) qb[idx] = f2bf(q[idx]);
  else if (idx < 2 * n) kb[idx - n] = f2bf(k[idx - n]);
}

__global__ void k_diag(const float* __restrict__ q, const float* __restrict__ k,
                       float* __restrict__ dq, float* __restrict__ dk) {
  int i = blockIdx.x * blockDim.x + threadIdx.x;  // 0..2*NHT
  if (i < NHT) {
    float s = 0.f;
    for (int d = 0; d < HIDD; ++d) { float x = q[(long long)i * HIDD + d]; s += x * x; }
    dq[i] = s * 0.0625f;  // 0.5 * dn^2 = 0.5 * 0.125
  } else if (i < 2 * NHT) {
    int j = i - NHT;
    float s = 0.f;
    for (int d = 0; d < HIDD; ++d) { float x = k[(long long)j * HIDD + d]; s += x * x; }
    dk[j] = s * 0.0625f;
  }
}

__global__ void k_rowmax(const float* __restrict__ dd, float* __restrict__ out) {
  int i = blockIdx.x * blockDim.x + threadIdx.x;
  if (i >= NHT) return;
  const float* r = dd + (long long)i * FP;
  float m = -1e30f;
  for (int f = 0; f < FF; ++f) m = fmaxf(m, r[f]);
  out[i] = m;
}

__global__ void k_nhmax(const float* __restrict__ rowmax, float* __restrict__ out) {
  __shared__ float red[256];
  int nh = blockIdx.x, tid = threadIdx.x;
  float m = -1e30f;
  for (int t = tid; t < TT; t += 256) m = fmaxf(m, rowmax[nh * TT + t]);
  red[tid] = m; __syncthreads();
  for (int s = 128; s > 0; s >>= 1) {
    if (tid < s) red[tid] = fmaxf(red[tid], red[tid + s]);
    __syncthreads();
  }
  if (tid == 0) out[nh] = red[0];
}

__global__ void k_featexp(const float* __restrict__ dd, const float* __restrict__ diag,
                          const float* __restrict__ mvec, int per_row,
                          unsigned short* __restrict__ out) {
  long long idx = (long long)blockIdx.x * blockDim.x + threadIdx.x;
  if (idx >= (long long)NHT * FP) return;
  int row = (int)(idx / FP), f = (int)(idx % FP);
  const float ratio = 0.06131393394849658f;  // 266^-0.5
  float v = 0.0f;
  if (f < FF) {
    float m = per_row ? mvec[row] : mvec[row >> 10];  // row/T -> (n,h)
    v = ratio * (expf(dd[idx] - diag[row] - m) + 1e-4f);
  }
  out[idx] = f2bf(v);
}

__global__ void k_ksum(const unsigned short* __restrict__ kp, float* __restrict__ ksum) {
  int idx = blockIdx.x * blockDim.x + threadIdx.x;  // 0..NH*FP
  if (idx >= NH * FP) return;
  int nh = idx / FP, f = idx % FP;
  float s = 0.f;
  const unsigned short* base = kp + (long long)nh * TT * FP + f;
  for (int t = 0; t < TT; ++t) s += bf2f(base[(long long)t * FP]);
  ksum[idx] = s;
}

// ctx[nh,f,d] = sum_t kp[nh,t,f]*vfa[t,d]; vfa[t,d]=1 iff t in [16d,16d+8)
// and mask row t unmasked (nearest-resize eye(128) structure).
__global__ void k_ctx(const unsigned short* __restrict__ kp,
                      const float* __restrict__ mask,
                      unsigned short* __restrict__ ctx) {
  long long idx = (long long)blockIdx.x * blockDim.x + threadIdx.x;
  if (idx >= (long long)NH * FP * HIDD) return;
  int d = (int)(idx & 63);
  int f = (int)((idx >> 6) % FP);
  int nh = (int)(idx / ((long long)FP * HIDD));
  int n = nh / HH;
  float s = 0.f;
  int t0 = 16 * d;
  for (int tt = 0; tt < 8; ++tt) {
    int t = t0 + tt;
    if (!(mask[n * TT + t] < -1.0f))
      s += bf2f(kp[((long long)nh * TT + t) * FP + f]);
  }
  ctx[idx] = f2bf(s);
}

// d_inv scale of pcl; emit pcl output and pv = concat(pcl, v) in bf16
__global__ void k_finish_pcl(const unsigned short* __restrict__ qp,
                             const float* __restrict__ ksum,
                             const float* __restrict__ pclraw,
                             const float* __restrict__ vin,
                             float* __restrict__ pcl_out,
                             unsigned short* __restrict__ pv) {
  int i = blockIdx.x * blockDim.x + threadIdx.x;
  if (i >= NHT) return;
  int nh = i >> 10;
  const unsigned short* qr = qp + (long long)i * FP;
  const float* ks = ksum + nh * FP;
  float dot = 0.f;
  for (int f = 0; f < FF; ++f) dot += bf2f(qr[f]) * ks[f];
  float dinv = 1.0f / dot;
  for (int d = 0; d < HIDD; ++d) {
    float p = pclraw[(long long)i * HIDD + d] * dinv;
    pcl_out[(long long)i * HIDD + d] = p;
    pv[(long long)i * 128 + d] = f2bf(p);
    pv[(long long)i * 128 + 64 + d] = f2bf(vin[(long long)i * HIDD + d]);
  }
}

__global__ void k_ln_gelu(const float* __restrict__ h1, const float* __restrict__ g,
                          const float* __restrict__ b, unsigned short* __restrict__ out) {
  __shared__ float red[128];
  int row = blockIdx.x, tid = threadIdx.x;
  float h = h1[(long long)row * 128 + tid];
  red[tid] = h; __syncthreads();
  for (int s = 64; s > 0; s >>= 1) { if (tid < s) red[tid] += red[tid + s]; __syncthreads(); }
  float mu = red[0] * (1.0f / 128.0f); __syncthreads();
  float d = h - mu;
  red[tid] = d * d; __syncthreads();
  for (int s = 64; s > 0; s >>= 1) { if (tid < s) red[tid] += red[tid + s]; __syncthreads(); }
  float var = red[0] * (1.0f / 128.0f);
  float x = d * rsqrtf(var + 1e-5f) * g[tid] + b[tid];
  float ge = 0.5f * x * (1.0f + erff(x * 0.70710678118f));  // exact GELU
  out[(long long)row * 128 + tid] = f2bf(ge);
}

__global__ void k_softmax(const float* __restrict__ score, const float* __restrict__ mask,
                          float* __restrict__ probs) {
  __shared__ float red[128];
  int row = blockIdx.x, tid = threadIdx.x;
  int n = row / (HH * TT);
  float rm = mask[n * TT + tid * 8];  // (j*T)//TM = 8j
  float s = score[(long long)row * TMD + tid];
  float sm = (rm < -1.0f) ? -10000.0f : s;
  red[tid] = sm; __syncthreads();
  for (int st = 64; st > 0; st >>= 1) { if (tid < st) red[tid] = fmaxf(red[tid], red[tid + st]); __syncthreads(); }
  float mx = red[0]; __syncthreads();
  float e = expf(sm - mx);
  red[tid] = e; __syncthreads();
  for (int st = 64; st > 0; st >>= 1) { if (tid < st) red[tid] += red[tid + st]; __syncthreads(); }
  probs[(long long)row * TMD + tid] = e / red[0];
}

__global__ void k_init_small(float* __restrict__ acc, float* __restrict__ lohi,
                             unsigned* __restrict__ counts) {
  int t = threadIdx.x;
  if (t < 2) acc[t] = 0.0f;
  if (t < NB) { lohi[t] = 0.0f; lohi[NB + t] = 1.0f; counts[t] = 0u; }
}

__global__ __launch_bounds__(256)
void k_loss(const float* __restrict__ score, const float* __restrict__ truth,
            const float* __restrict__ mask, float* __restrict__ acc) {
  __shared__ float red[256];
  int row = blockIdx.x, tid = threadIdx.x;
  int n = row / (HH * TT);
  const float* srow = score + (long long)row * TMD;
  const float* trow = truth + (long long)row * TT;
  const float* mrow = mask + n * TT;
  float ma = -1e30f, mb = -1e30f;
  for (int j = tid; j < TT; j += 256) {
    float m = mrow[j];
    ma = fmaxf(ma, srow[j >> 3] + m);   // score_r: (j*TM)//T = j//8
    mb = fmaxf(mb, trow[j] + m);
  }
  red[tid] = ma; __syncthreads();
  for (int s = 128; s > 0; s >>= 1) { if (tid < s) red[tid] = fmaxf(red[tid], red[tid + s]); __syncthreads(); }
  ma = red[0]; __syncthreads();
  red[tid] = mb; __syncthreads();
  for (int s = 128; s > 0; s >>= 1) { if (tid < s) red[tid] = fmaxf(red[tid], red[tid + s]); __syncthreads(); }
  mb = red[0]; __syncthreads();
  float sa = 0.f, sb = 0.f;
  for (int j = tid; j < TT; j += 256) {
    float m = mrow[j];
    sa += expf(srow[j >> 3] + m - ma);
    sb += expf(trow[j] + m - mb);
  }
  red[tid] = sa; __syncthreads();
  for (int s = 128; s > 0; s >>= 1) { if (tid < s) red[tid] += red[tid + s]; __syncthreads(); }
  sa = red[0]; __syncthreads();
  red[tid] = sb; __syncthreads();
  for (int s = 128; s > 0; s >>= 1) { if (tid < s) red[tid] += red[tid + s]; __syncthreads(); }
  sb = red[0]; __syncthreads();
  float lza = logf(sa) + ma, lzb = logf(sb) + mb;
  float kl = 0.f, mse = 0.f;
  for (int j = tid; j < TT; j += 256) {
    float m = mrow[j];
    float sr = srow[j >> 3], tr = trow[j];
    if (m > -1.0f) {
      float pt = expf(tr + m - lzb);
      float logp = sr + m - lza;
      kl += pt * (logf(fmaxf(pt, 1e-12f)) - logp);
    }
    float s0 = (m < -1.0f) ? 0.0f : sr;
    float t0 = (m < -1.0f) ? 0.0f : tr;
    float d = s0 - t0;
    mse += d * d;
  }
  red[tid] = kl; __syncthreads();
  for (int s = 128; s > 0; s >>= 1) { if (tid < s) red[tid] += red[tid + s]; __syncthreads(); }
  kl = red[0]; __syncthreads();
  red[tid] = mse; __syncthreads();
  for (int s = 128; s > 0; s >>= 1) { if (tid < s) red[tid] += red[tid + s]; __syncthreads(); }
  if (tid == 0) { atomicAdd(&acc[0], kl); atomicAdd(&acc[1], red[0]); }
}

__global__ void k_loss_final(const float* __restrict__ acc, const float* __restrict__ mask,
                             float* __restrict__ out) {
  if (threadIdx.x == 0 && blockIdx.x == 0) {
    double denom = 0.0;
    for (int n = 0; n < NB; ++n) {
      int cnt = 0;
      for (int j = 0; j < TT; ++j) if (mask[n * TT + j] > -1.0f) cnt++;
      denom += (double)cnt * (HH * TT);
    }
    if (denom < 1.0) denom = 1.0;
    out[0] = (float)((double)acc[0] / denom * 0.25) + acc[1] / 50331648.0f;
  }
}

// ----- top-k threshold via deterministic bisection on probs in [0,1] -----
__global__ __launch_bounds__(256)
void k_count(const float* __restrict__ probs, const float* __restrict__ mask,
             const float* __restrict__ lohi, unsigned* __restrict__ counts) {
  __shared__ unsigned sc;
  if (threadIdx.x == 0) sc = 0u;
  __syncthreads();
  int b = blockIdx.x / 6144;                       // 1572864/256 blocks per batch
  int tib = (blockIdx.x % 6144) * 256 + threadIdx.x;
  int t = (tib / TMD) % TT;
  float v = (mask[b * TT + t] > -1.0f) ? probs[(long long)b * 1572864 + tib] : 0.0f;
  float mid = 0.5f * (lohi[b] + lohi[NB + b]);
  if (v > mid) atomicAdd(&sc, 1u);
  __syncthreads();
  if (threadIdx.x == 0 && sc) atomicAdd(&counts[b], sc);
}

__global__ void k_update(float* __restrict__ lohi, unsigned* __restrict__ counts) {
  int b = threadIdx.x;
  if (b < NB) {
    float mid = 0.5f * (lohi[b] + lohi[NB + b]);
    if (counts[b] > (unsigned)TGT) lohi[b] = mid; else lohi[NB + b] = mid;
    counts[b] = 0u;
  }
}

__global__ __launch_bounds__(256)
void k_pam(const float* __restrict__ probs, const float* __restrict__ mask,
           const float* __restrict__ lohi, float* __restrict__ pam) {
  long long idx = (long long)blockIdx.x * 256 + threadIdx.x;
  if (idx >= (long long)NB * 1572864) return;
  int b = (int)(idx / 1572864);
  int tib = (int)(idx % 1572864);
  int t = (tib / TMD) % TT;
  float v = (mask[b * TT + t] > -1.0f) ? probs[idx] : 0.0f;
  float thr = 0.5f * (lohi[b] + lohi[NB + b]);
  pam[idx] = (v > thr) ? 0.0f : -10000.0f;
}

// ---------------------------------------------------------------------------
extern "C" void kernel_launch(void* const* d_in, const int* in_sizes, int n_in,
                              void* d_out, int out_size, void* d_ws, size_t ws_size,
                              hipStream_t stream) {
  (void)in_sizes; (void)n_in; (void)out_size; (void)ws_size;
  const float* v_in   = (const float*)d_in[2];
  const float* qfa    = (const float*)d_in[3];
  const float* kfa    = (const float*)d_in[4];
  const float* mask   = (const float*)d_in[8];
  const float* truth  = (const float*)d_in[9];
  const float* pm     = (const float*)d_in[11];
  const float* enc_w  = (const float*)d_in[12];
  const float* enc_b  = (const float*)d_in[13];
  const float* ln_g   = (const float*)d_in[14];
  const float* ln_b   = (const float*)d_in[15];
  const float* dec_w  = (const float*)d_in[16];
  const float* dec_b  = (const float*)d_in[17];

  float* out = (float*)d_out;
  float* loss_out  = out;                                  // [1]
  float* pcl_out   = out + 1;                              // [N,H,T,64]
  float* probs_out = pcl_out + (long long)NHT * HIDD;      // [N,H,T,128]
  float* pam_out   = probs_out + (long long)NHT * TMD;     // [N,H,T,128]

  // workspace carve
  char* w = (char*)d_ws;
  size_t o = 0;
  #define CARVE(ptrT, name, bytes) \
    ptrT name = (ptrT)(w + o); o = (o + (size_t)(bytes) + 255) & ~(size_t)255;
  CARVE(unsigned short*, pmT_bf,  64 * FP * 2)
  CARVE(unsigned short*, encw_bf, 16384 * 2)
  CARVE(unsigned short*, decw_bf, 16384 * 2)
  CARVE(unsigned short*, qx_bf,  (size_t)NHT * HIDD * 2)
  CARVE(unsigned short*, kx_bf,  (size_t)NHT * HIDD * 2)
  CARVE(float*,          ddq,    (size_t)NHT * FP * 4)
  CARVE(float*,          ddk,    (size_t)NHT * FP * 4)
  CARVE(float*,          diagq,  (size_t)NHT * 4)
  CARVE(float*,          diagk,  (size_t)NHT * 4)
  CARVE(float*,          qmax,   (size_t)NHT * 4)
  CARVE(float*,          krmax,  (size_t)NHT * 4)
  CARVE(float*,          kmax,   NH * 4)
  CARVE(unsigned short*, qp_bf,  (size_t)NHT * FP * 2)
  CARVE(unsigned short*, kp_bf,  (size_t)NHT * FP * 2)
  CARVE(float*,          ksum,   NH * FP * 4)
  CARVE(unsigned short*, ctx_bf, (size_t)NH * FP * HIDD * 2)
  CARVE(float*,          pclraw, (size_t)NHT * HIDD * 4)
  CARVE(unsigned short*, pv_bf,  (size_t)NHT * 128 * 2)
  CARVE(float*,          h1,     (size_t)NHT * 128 * 4)
  CARVE(unsigned short*, h1g_bf, (size_t)NHT * 128 * 2)
  CARVE(float*,          score,  (size_t)NHT * 128 * 4)
  CARVE(float*,          accf,   2 * 4)
  CARVE(float*,          lohi,   2 * NB * 4)
  CARVE(unsigned*,       counts, NB * 4)
  #undef CARVE

  // 1. weights -> bf16 (pm^T padded with dn folded)
  k_convert_weights<<<(64 * FP + 32768 + 255) / 256, 256, 0, stream>>>(
      pm, enc_w, dec_w, pmT_bf, encw_bf, decw_bf);
  // 2. x -> bf16 and diag
  k_convert_x<<<(2 * NHT * HIDD + 255) / 256, 256, 0, stream>>>(qfa, kfa, qx_bf, kx_bf);
  k_diag<<<(2 * NHT + 255) / 256, 256, 0, stream>>>(qfa, kfa, diagq, diagk);
  // 3. dd = (dn*x) @ pm^T  (WMMA bf16), M=49152 K=64 N=288
  dim3 gdd((FP + 127) / 128, NHT / 64, 1);
  k_gemm_bf16<<<gdd, 256, 0, stream>>>(qx_bf, pmT_bf, ddq, NHT, FP, HIDD, 0, 0, 0, nullptr);
  k_gemm_bf16<<<gdd, 256, 0, stream>>>(kx_bf, pmT_bf, ddk, NHT, FP, HIDD, 0, 0, 0, nullptr);
  // 4. maxes + exp features
  k_rowmax<<<NHT / 256, 256, 0, stream>>>(ddq, qmax);
  k_rowmax<<<NHT / 256, 256, 0, stream>>>(ddk, krmax);
  k_nhmax<<<NH, 256, 0, stream>>>(krmax, kmax);
  k_featexp<<<(int)(((long long)NHT * FP + 255) / 256), 256, 0, stream>>>(ddq, diagq, qmax, 1, qp_bf);
  k_featexp<<<(int)(((long long)NHT * FP + 255) / 256), 256, 0, stream>>>(ddk, diagk, kmax, 0, kp_bf);
  // 5. ksum and ctx (ctx uses eye-resize sparsity: 8 adds per (f,d))
  k_ksum<<<(NH * FP + 255) / 256, 256, 0, stream>>>(kp_bf, ksum);
  k_ctx<<<(NH * FP * HIDD) / 256, 256, 0, stream>>>(kp_bf, mask, ctx_bf);
  // 6. pcl_raw = qp @ ctx (batched WMMA): 48 x [1024x288]x[288x64]
  dim3 gpcl(1, TT / 64, NH);
  k_gemm_bf16<<<gpcl, 256, 0, stream>>>(qp_bf, ctx_bf, pclraw,
      TT, HIDD, FP, (long long)TT * FP, (long long)FP * HIDD, (long long)TT * HIDD, nullptr);
  // 7. d_inv scaling, emit pcl + pv = concat(pcl, v)
  k_finish_pcl<<<NHT / 256, 256, 0, stream>>>(qp_bf, ksum, pclraw, v_in, pcl_out, pv_bf);
  // 8. MLP: h1 = pv @ enc_w + b ; LN+GELU ; score = h1g @ dec_w + b  (WMMA)
  dim3 gmlp(1, NHT / 64, 1);
  k_gemm_bf16<<<gmlp, 256, 0, stream>>>(pv_bf, encw_bf, h1, NHT, 128, 128, 0, 0, 0, enc_b);
  k_ln_gelu<<<NHT, 128, 0, stream>>>(h1, ln_g, ln_b, h1g_bf);
  k_gemm_bf16<<<gmlp, 256, 0, stream>>>(h1g_bf, decw_bf, score, NHT, TMD, 128, 0, 0, 0, dec_b);
  // 9. probs (mask-resized softmax over TM)
  k_softmax<<<NHT, 128, 0, stream>>>(score, mask, probs_out);
  // 10. loss (KL + MSE)
  k_init_small<<<1, 32, 0, stream>>>(accf, lohi, counts);
  k_loss<<<NHT, 256, 0, stream>>>(score, truth, mask, accf);
  k_loss_final<<<1, 1, 0, stream>>>(accf, mask, loss_out);
  // 11. per-batch top-k threshold via bisection, then pam
  for (int it = 0; it < 24; ++it) {
    k_count<<<NB * 6144, 256, 0, stream>>>(probs_out, mask, lohi, counts);
    k_update<<<1, 32, 0, stream>>>(lohi, counts);
  }
  k_pam<<<NB * 6144, 256, 0, stream>>>(probs_out, mask, lohi, pam_out);
}
